// MolecularDiscriminator_42838003810623
// MI455X (gfx1250) — matrix operand order repaired and checked by reading, not compiled, not run on hardware
//
#include <hip/hip_runtime.h>
#include <hip/hip_bf16.h>

// ---------------- CDNA5 WMMA helpers (wave32, 16x16x32 f16 -> f32) ----------

typedef __attribute__((ext_vector_type(16))) _Float16 v16h;
typedef __attribute__((ext_vector_type(8)))  float    v8f;

__device__ __forceinline__ v8f wmma_f16(v16h a, v16h b, v8f c) {
  // D = A x B + C ; args: (neg_a, A, neg_b, B, c_mod, C, reuse_a, reuse_b)
  return __builtin_amdgcn_wmma_f32_16x16x32_f16(false, a, false, b, (short)0, c,
                                                false, false);
}

__device__ __forceinline__ float silu(float x) {
  return x / (1.0f + __expf(-x));
}

// A-fragment (16x32 MxK, f16) swizzled storage: element (m, k) of a 32-wide
// K-block lives in lane = m + 16*((k>>3)&1), at half-index:
//   e = 2*( ((k>>4)&1)*4 + ((k&7)>>1) ) + (k&1)
// A fragment for (kk, lane) is 16 contiguous halves (32 bytes) -> 2x b128 load.
__device__ __forceinline__ int afrag_lane(int m, int k) {
  return m + 16 * ((k >> 3) & 1);
}
__device__ __forceinline__ int afrag_elem(int k) {
  return 2 * ((((k >> 4) & 1) << 2) + ((k & 7) >> 1)) + (k & 1);
}
// Store one value into a swizzled A tile (tile = 16 rows, K blocks of 32).
__device__ __forceinline__ void store_a_sw(_Float16* tile_base, int m, int c,
                                           _Float16 v) {
  int kk = c >> 5, k = c & 31;
  tile_base[(kk * 32 + afrag_lane(m, k)) * 16 + afrag_elem(k)] = v;
}
__device__ __forceinline__ v16h load_afrag_sw(const _Float16* tile_base, int kk,
                                              int lane) {
  return *(const v16h*)(tile_base + (kk * 32 + lane) * 16);
}

// Packed B-fragment buffers: fragment `f` for (kk, nt) stored as
// 32 lanes x 16 halves contiguous (1 KB per fragment).
#define FRAG_H 512  // halves per fragment
__device__ __forceinline__ v16h load_bfrag_pk(const _Float16* pk, int frag,
                                              int lane) {
  return *(const v16h*)(pk + frag * FRAG_H + lane * 16);
}

// ---------------- Problem constants ----------------------------------------
#define BSZ   128
#define NN    64
#define IN_NF 256
#define HF    64
#define ROWS  (BSZ * NN)          // 8192 node rows
#define INV_NORM 0.01f            // 1/100

// ---------------- Weight packing kernels (run once per call) ----------------
// Generic: row-major [K x ldn] f32 -> B-fragments, frag = kk*nnt + nt.
__global__ __launch_bounds__(32)
void k_pack_b(const float* __restrict__ w, int ldn, int nnt,
              _Float16* __restrict__ dst) {
  int frag = blockIdx.x;
  int lane = threadIdx.x;
  int kk = frag / nnt, nt = frag % nnt;
  const float* src = w + kk * 32 * ldn + nt * 16;
  int n = lane & 15, khalf = lane >> 4;
  v16h b;
#pragma unroll
  for (int p = 0; p < 8; ++p) {
    int k = khalf * 8 + 2 * (p & 3) + 16 * (p >> 2);
    b[2 * p]     = (_Float16)src[k * ldn + n];
    b[2 * p + 1] = (_Float16)src[(k + 1) * ldn + n];
  }
  *(v16h*)(dst + frag * FRAG_H + lane * 16) = b;
}

// e_w1 [129 x 64]: build virtual [64 x 128] W'[k][c] = e_w1[(c<64?k:64+k)][c&63]
// (h_i weights in cols 0..63, h_j weights in cols 64..127). frag = kk*8 + nt.
__global__ __launch_bounds__(32)
void k_pack_b_split(const float* __restrict__ w, _Float16* __restrict__ dst) {
  int frag = blockIdx.x;
  int lane = threadIdx.x;
  int kk = frag >> 3, nt = frag & 7;
  int n = lane & 15, khalf = lane >> 4;
  int c = nt * 16 + n;
  int roff = (c < HF) ? 0 : HF;
  int col = c & 63;
  v16h b;
#pragma unroll
  for (int p = 0; p < 8; ++p) {
    int k = kk * 32 + khalf * 8 + 2 * (p & 3) + 16 * (p >> 2);
    b[2 * p]     = (_Float16)w[(roff + k) * HF + col];
    b[2 * p + 1] = (_Float16)w[(roff + k + 1) * HF + col];
  }
  *(v16h*)(dst + frag * FRAG_H + lane * 16) = b;
}

// ---------------- Kernel 1: embedding  h = (h_in*mask) @ emb_w + emb_b -----
// grid 512 (16-row tiles), block 128 (4 waves; wave w -> output cols 16w..)
__global__ __launch_bounds__(128)
void k_embed(const float* __restrict__ h_in, const float* __restrict__ node_mask,
             const _Float16* __restrict__ pk_emb, const float* __restrict__ emb_b,
             float* __restrict__ h_f32, _Float16* __restrict__ h16_sw) {
  __shared__ alignas(32) _Float16 sA[16 * IN_NF];  // swizzled A tile, 8 KB
  int tile = blockIdx.x;
  int t = threadIdx.x;
  int row0 = tile * 16;

  // Stage + mask + convert + swizzle: coalesced float4 reads.
#pragma unroll
  for (int e4 = t; e4 < 16 * IN_NF / 4; e4 += 128) {
    int m = e4 >> 6;            // 64 float4 per row
    int c4 = (e4 & 63) * 4;
    float mask = node_mask[row0 + m];
    float4 v = *(const float4*)&h_in[(row0 + m) * IN_NF + c4];
    store_a_sw(sA, m, c4 + 0, (_Float16)(v.x * mask));
    store_a_sw(sA, m, c4 + 1, (_Float16)(v.y * mask));
    store_a_sw(sA, m, c4 + 2, (_Float16)(v.z * mask));
    store_a_sw(sA, m, c4 + 3, (_Float16)(v.w * mask));
  }
  __syncthreads();

  int wave = t >> 5, lane = t & 31;
  v8f acc = {};
#pragma unroll
  for (int kk = 0; kk < 8; ++kk) {
    __builtin_prefetch(pk_emb + ((kk + 1) & 7) * 4 * FRAG_H, 0, 1);
    v16h a = load_afrag_sw(sA, kk, lane);
    v16h b = load_bfrag_pk(pk_emb, kk * 4 + wave, lane);
    acc = wmma_f16(a, b, acc);
  }
  int c = wave * 16 + (lane & 15);
  float bias = emb_b[c];
  _Float16* htile = h16_sw + tile * 1024;  // 2 K-blocks * 32 lanes * 16 halves
#pragma unroll
  for (int r = 0; r < 8; ++r) {
    int m = r + 8 * (lane >> 4);
    float v = acc[r] + bias;
    h_f32[(row0 + m) * HF + c] = v;
    store_a_sw(htile, m, c, (_Float16)v);
  }
}

// ---------------- Kernel 2: pre-edge  AB = h @ [w1_src | w1_dst] (+b1 on A) -
// grid 512, block 256 (8 waves; wave w -> output cols 16w of the 128)
__global__ __launch_bounds__(256)
void k_preedge(const _Float16* __restrict__ h16_sw,
               const _Float16* __restrict__ pk_pre, const float* __restrict__ e_b1,
               float* __restrict__ AB) {
  int tile = blockIdx.x;
  int wave = threadIdx.x >> 5;
  int lane = threadIdx.x & 31;
  int row0 = tile * 16;
  const _Float16* htile = h16_sw + tile * 1024;

  v8f acc = {};
#pragma unroll
  for (int kk = 0; kk < 2; ++kk) {
    v16h a = load_afrag_sw(htile, kk, lane);
    v16h b = load_bfrag_pk(pk_pre, kk * 8 + wave, lane);
    acc = wmma_f16(a, b, acc);
  }
  int c = wave * 16 + (lane & 15);
  float bias = (c < HF) ? e_b1[c] : 0.0f;  // bake b1 into the A half (added once)
#pragma unroll
  for (int r = 0; r < 8; ++r) {
    int row = row0 + r + 8 * (lane >> 4);
    AB[row * 128 + c] = acc[r] + bias;
  }
}

// ---------------- Kernel 3: edge MLP second matmul + aggregation -----------
// one workgroup per (b, i): e1[j][c] = silu(A_i[c] + B_j[c] + em*w1[128][c] + b1)
// mij = silu(e1 @ w2 + b2) * em ; agg_i = sum_j mij / 100
// grid 8192, block 256 (8 waves; 16 output tiles of 16x16 -> 2 per wave)
__global__ __launch_bounds__(256)
void k_edge(const float* __restrict__ AB, const float* __restrict__ e_w1,
            const _Float16* __restrict__ pk_ew2, const float* __restrict__ e_b2,
            const float* __restrict__ edge_mask, _Float16* __restrict__ agg_sw) {
  __shared__ float sA[NN];                          // A_i + b1
  __shared__ float sEm[NN];                         // edge mask row
  __shared__ float sAgg[NN];                        // aggregation accumulator
  __shared__ alignas(32) _Float16 sE1[NN * HF];     // swizzled A tiles: [mt][kk][lane][16]

  int g = blockIdx.x;  // node index = b*64 + i
  int t = threadIdx.x;
  if (t < NN) {
    sA[t]   = AB[g * 128 + t];
    sEm[t]  = edge_mask[g * NN + t];
    sAgg[t] = 0.0f;
  }
  __syncthreads();

  const float* w1r128 = e_w1 + 128 * HF;  // em-coefficient row of e_w1
  int bbase = (g >> 6) * NN;              // first node row of this batch
#pragma unroll
  for (int e4 = t; e4 < NN * HF / 4; e4 += 256) {
    int j = e4 >> 4;            // 16 float4 per row of 64 cols
    int c4 = (e4 & 15) * 4;
    float4 bj = *(const float4*)&AB[(bbase + j) * 128 + HF + c4];
    float4 wr = *(const float4*)&w1r128[c4];
    float4 ai = *(const float4*)&sA[c4];
    float em = sEm[j];
    int mt = j >> 4, m = j & 15;
    _Float16* tb = sE1 + mt * 1024;
    store_a_sw(tb, m, c4 + 0, (_Float16)silu(ai.x + bj.x + em * wr.x));
    store_a_sw(tb, m, c4 + 1, (_Float16)silu(ai.y + bj.y + em * wr.y));
    store_a_sw(tb, m, c4 + 2, (_Float16)silu(ai.z + bj.z + em * wr.z));
    store_a_sw(tb, m, c4 + 3, (_Float16)silu(ai.w + bj.w + em * wr.w));
  }
  __syncthreads();

  int wave = t >> 5, lane = t & 31;
#pragma unroll
  for (int sub = 0; sub < 2; ++sub) {
    int id = wave * 2 + sub;
    int mt = id >> 2, nt = id & 3;  // 4 M-tiles (j) x 4 N-tiles (c)
    v8f acc = {};
#pragma unroll
    for (int kk = 0; kk < 2; ++kk) {
      v16h a = load_afrag_sw(sE1 + mt * 1024, kk, lane);
      v16h b = load_bfrag_pk(pk_ew2, kk * 4 + nt, lane);
      acc = wmma_f16(a, b, acc);
    }
    int c = nt * 16 + (lane & 15);
    float b2v = e_b2[c];
    float part = 0.0f;
#pragma unroll
    for (int r = 0; r < 8; ++r) {
      int j = mt * 16 + r + 8 * (lane >> 4);
      part += silu(acc[r] + b2v) * sEm[j];
    }
    atomicAdd(&sAgg[c], part);  // ds_add_f32
  }
  __syncthreads();
  if (t < NN) {
    // agg row g, col t -> swizzled A layout for the node-update kernel
    int tile = g >> 4, m = g & 15;
    store_a_sw(agg_sw + tile * 1024, m, t, (_Float16)(sAgg[t] * INV_NORM));
  }
}

// ---------------- Kernel 4: node update ------------------------------------
// t = silu([h|agg] @ nw1 + nb1); h = (h + t @ nw2 + nb2) * node_mask
// grid 512, block 128 (4 waves)
__global__ __launch_bounds__(128)
void k_node(const _Float16* __restrict__ agg_sw,
            const _Float16* __restrict__ pk_nw1, const float* __restrict__ n_b1,
            const _Float16* __restrict__ pk_nw2, const float* __restrict__ n_b2,
            const float* __restrict__ node_mask,
            float* __restrict__ h_f32, _Float16* __restrict__ h16_sw) {
  __shared__ alignas(32) _Float16 sT[16 * HF];  // swizzled, 2 KB
  int tile = blockIdx.x;
  int wave = threadIdx.x >> 5;
  int lane = threadIdx.x & 31;
  int row0 = tile * 16;
  _Float16* htile = h16_sw + tile * 1024;
  const _Float16* atile = agg_sw + tile * 1024;

  v8f acc = {};
#pragma unroll
  for (int kk = 0; kk < 4; ++kk) {
    // K 0..63 from h, K 64..127 from agg — no concat buffer needed.
    v16h a = (kk < 2) ? load_afrag_sw(htile, kk, lane)
                      : load_afrag_sw(atile, kk - 2, lane);
    v16h b = load_bfrag_pk(pk_nw1, kk * 4 + wave, lane);
    acc = wmma_f16(a, b, acc);
  }
  int c = wave * 16 + (lane & 15);
  float b1v = n_b1[c];
#pragma unroll
  for (int r = 0; r < 8; ++r) {
    int m = r + 8 * (lane >> 4);
    store_a_sw(sT, m, c, (_Float16)silu(acc[r] + b1v));
  }
  __syncthreads();

  v8f acc2 = {};
#pragma unroll
  for (int kk = 0; kk < 2; ++kk) {
    v16h a = load_afrag_sw(sT, kk, lane);
    v16h b = load_bfrag_pk(pk_nw2, kk * 4 + wave, lane);
    acc2 = wmma_f16(a, b, acc2);
  }
  float b2v = n_b2[c];
#pragma unroll
  for (int r = 0; r < 8; ++r) {
    int m = r + 8 * (lane >> 4);
    int row = row0 + m;
    float v = (h_f32[row * HF + c] + acc2[r] + b2v) * node_mask[row];
    h_f32[row * HF + c] = v;
    store_a_sw(htile, m, c, (_Float16)v);
  }
}

// ---------------- Kernel 5: out-proj + masked mean pool + head MLP ---------
// grid 128 (one per batch), block 256 (8 waves; 64 tiles -> 8 per wave)
__global__ __launch_bounds__(256)
void k_head(const _Float16* __restrict__ h16_sw, const _Float16* __restrict__ pk_out,
            const float* __restrict__ out_b, const float* __restrict__ node_mask,
            const float* __restrict__ mlp_w1, const float* __restrict__ mlp_b1,
            const float* __restrict__ mlp_w2, const float* __restrict__ mlp_b2,
            float* __restrict__ out) {
  __shared__ float sPool[IN_NF];
  __shared__ float sY[IN_NF];
  __shared__ float sAtom;
  int b = blockIdx.x;
  int t = threadIdx.x;
  int wave = t >> 5, lane = t & 31;

  sPool[t] = 0.0f;
  if (t == 0) {
    float a = 0.0f;
    for (int j = 0; j < NN; ++j) a += node_mask[b * NN + j];
    sAtom = a;
  }
  __syncthreads();

#pragma unroll
  for (int s = 0; s < 8; ++s) {
    int id = wave * 8 + s;
    int mt = id >> 4, nt = id & 15;  // 4 M-tiles (nodes) x 16 N-tiles (256 cols)
    v8f acc = {};
#pragma unroll
    for (int kk = 0; kk < 2; ++kk) {
      v16h a = load_afrag_sw(h16_sw + (b * 4 + mt) * 1024, kk, lane);
      v16h w = load_bfrag_pk(pk_out, kk * 16 + nt, lane);
      acc = wmma_f16(a, w, acc);
    }
    int c = nt * 16 + (lane & 15);
    float bias = out_b[c];
    float part = 0.0f;
#pragma unroll
    for (int r = 0; r < 8; ++r) {
      int j = mt * 16 + r + 8 * (lane >> 4);
      part += (acc[r] + bias) * node_mask[b * NN + j];
    }
    atomicAdd(&sPool[c], part);
  }
  __syncthreads();

  float x0 = sPool[t] / sAtom;  // pooled (mean over present atoms)
  sPool[t] = x0;
  __syncthreads();

  // head MLP layer 1: y[t] = silu(sum_k pooled[k] * mlp_w1[k][t] + b1[t])
  float acc1 = mlp_b1[t];
  for (int k = 0; k < IN_NF; ++k) acc1 += sPool[k] * mlp_w1[k * IN_NF + t];
  float yv = silu(acc1);

  // head MLP layer 2 (256 -> 1) + log(sigmoid)
  sY[t] = yv * mlp_w2[t];
  __syncthreads();
  for (int off = 128; off > 0; off >>= 1) {
    if (t < off) sY[t] += sY[t + off];
    __syncthreads();
  }
  if (t == 0) {
    float z = sY[0] + mlp_b2[0];
    out[b] = -log1pf(__expf(-z));  // log(sigmoid(z))
  }
}

// ---------------- Host launch ----------------------------------------------
extern "C" void kernel_launch(void* const* d_in, const int* in_sizes, int n_in,
                              void* d_out, int out_size, void* d_ws, size_t ws_size,
                              hipStream_t stream) {
  (void)in_sizes; (void)n_in; (void)out_size; (void)ws_size;
  const float* h_in      = (const float*)d_in[0];
  const float* node_mask = (const float*)d_in[1];
  const float* edge_mask = (const float*)d_in[2];
  const float* emb_w     = (const float*)d_in[3];
  const float* emb_b     = (const float*)d_in[4];
  const float* ew1[2] = {(const float*)d_in[5],  (const float*)d_in[13]};
  const float* eb1[2] = {(const float*)d_in[6],  (const float*)d_in[14]};
  const float* ew2[2] = {(const float*)d_in[7],  (const float*)d_in[15]};
  const float* eb2[2] = {(const float*)d_in[8],  (const float*)d_in[16]};
  const float* nw1[2] = {(const float*)d_in[9],  (const float*)d_in[17]};
  const float* nb1[2] = {(const float*)d_in[10], (const float*)d_in[18]};
  const float* nw2[2] = {(const float*)d_in[11], (const float*)d_in[19]};
  const float* nb2[2] = {(const float*)d_in[12], (const float*)d_in[20]};
  const float* out_w  = (const float*)d_in[21];
  const float* out_b  = (const float*)d_in[22];
  const float* mlp_w1 = (const float*)d_in[23];
  const float* mlp_b1 = (const float*)d_in[24];
  const float* mlp_w2 = (const float*)d_in[25];
  const float* mlp_b2 = (const float*)d_in[26];

  char* ws = (char*)d_ws;
  float*    h32    = (float*)ws;                      // [8192][64] f32: 2 MB
  _Float16* h16_sw = (_Float16*)(ws + (2u << 20));    // swizzled A tiles: 1 MB
  float*    AB     = (float*)(ws + (3u << 20));       // [8192][128] f32: 4 MB
  _Float16* agg_sw = (_Float16*)(ws + (7u << 20));    // swizzled A tiles: 1 MB

  // Packed f16 B-fragment weights (1 KB per 16x16x32 fragment).
  char* pkbase = ws + (8u << 20);
  _Float16* pk_emb    = (_Float16*)(pkbase);                 // 32 frags
  _Float16* pk_pre[2] = {(_Float16*)(pkbase + 32 * 1024),    // 16 frags each
                         (_Float16*)(pkbase + 48 * 1024)};
  _Float16* pk_ew2[2] = {(_Float16*)(pkbase + 64 * 1024),    // 8 frags each
                         (_Float16*)(pkbase + 72 * 1024)};
  _Float16* pk_nw1[2] = {(_Float16*)(pkbase + 80 * 1024),    // 16 frags each
                         (_Float16*)(pkbase + 96 * 1024)};
  _Float16* pk_nw2[2] = {(_Float16*)(pkbase + 112 * 1024),   // 8 frags each
                         (_Float16*)(pkbase + 120 * 1024)};
  _Float16* pk_out    = (_Float16*)(pkbase + 128 * 1024);    // 32 frags

  // Pack weights (tiny; all L2-resident afterwards).
  k_pack_b<<<32, 32, 0, stream>>>(emb_w, HF, 4, pk_emb);
  for (int l = 0; l < 2; ++l) {
    k_pack_b_split<<<16, 32, 0, stream>>>(ew1[l], pk_pre[l]);
    k_pack_b<<<8, 32, 0, stream>>>(ew2[l], HF, 4, pk_ew2[l]);
    k_pack_b<<<16, 32, 0, stream>>>(nw1[l], HF, 4, pk_nw1[l]);
    k_pack_b<<<8, 32, 0, stream>>>(nw2[l], HF, 4, pk_nw2[l]);
  }
  k_pack_b<<<32, 32, 0, stream>>>(out_w, IN_NF, 16, pk_out);

  k_embed<<<ROWS / 16, 128, 0, stream>>>(h_in, node_mask, pk_emb, emb_b, h32, h16_sw);

  for (int l = 0; l < 2; ++l) {
    k_preedge<<<ROWS / 16, 256, 0, stream>>>(h16_sw, pk_pre[l], eb1[l], AB);
    k_edge<<<ROWS, 256, 0, stream>>>(AB, ew1[l], pk_ew2[l], eb2[l], edge_mask, agg_sw);
    k_node<<<ROWS / 16, 128, 0, stream>>>(agg_sw, pk_nw1[l], nb1[l], pk_nw2[l],
                                          nb2[l], node_mask, h32, h16_sw);
  }

  k_head<<<BSZ, 256, 0, stream>>>(h16_sw, pk_out, out_b, node_mask,
                                  mlp_w1, mlp_b1, mlp_w2, mlp_b2, (float*)d_out);
}
